// RoutedMoE_32710470926991
// MI455X (gfx1250) — matrix operand order repaired and checked by reading, not compiled, not run on hardware
//
#include <hip/hip_runtime.h>
#include <hip/hip_bf16.h>

// MI455X / gfx1250 routed-MoE.
// - bf16 WMMA (v_wmma_f32_16x16x32_bf16) for all three GEMMs, f32 accumulate.
// - Routed (top-2) instead of dense-masked: 4x fewer FLOPs than reference path.
// - A-tiles staged with GLOBAL_LOAD_ASYNC_TO_LDS_B128 (per-lane gather addresses),
//   double-buffered, synchronized with s_wait_asynccnt (ASYNCcnt, in-order).

typedef __attribute__((ext_vector_type(16))) __bf16 v16bf;
typedef __attribute__((ext_vector_type(8)))  float  v8f;

#define BB   4
#define SS   2048
#define DD   2048
#define MM   2048
#define EE   8
#define NTOK (BB * SS)      // 8192 tokens
#define NPAIR (NTOK * 2)    // 16384 (token, expert) pairs
#define LISTCAP NPAIR

__device__ __forceinline__ unsigned short f2bf(float f) {
  unsigned u = __builtin_bit_cast(unsigned, f);
  u += 0x7FFFu + ((u >> 16) & 1u);   // round-to-nearest-even
  return (unsigned short)(u >> 16);
}

union Frag { uint4 q[2]; v16bf v; };

__device__ __forceinline__ unsigned lds_off(const void* p) {
  // generic LDS pointer: low 32 bits are the wave-relative LDS byte address
  return (unsigned)(unsigned long long)p;
}

// async copy of 16B per lane, global -> LDS (tracked by ASYNCcnt)
__device__ __forceinline__ void a_async16(unsigned lds_byte, const void* g) {
  asm volatile("global_load_async_to_lds_b128 %0, %1, off"
               :: "v"(lds_byte), "v"(g) : "memory");
}

// ---------------------------------------------------------------- init
__global__ void k_init(int* __restrict__ counts) {
  if (threadIdx.x < EE) counts[threadIdx.x] = 0;
}

// ---------------------------------------------------------------- gate + routing + x->bf16
__global__ __launch_bounds__(256) void k_gate(
    const float* __restrict__ x, const float* __restrict__ gk,
    unsigned short* __restrict__ xbf, float* __restrict__ pw,
    int* __restrict__ counts, int* __restrict__ lists) {
  const int t = blockIdx.x;
  const float* xr = x + (size_t)t * DD;
  float p[EE];
#pragma unroll
  for (int e = 0; e < EE; e++) p[e] = 0.f;

  for (int d = threadIdx.x; d < DD; d += 256) {
    float xv = xr[d];
    xbf[(size_t)t * DD + d] = f2bf(xv);
    const float* g = gk + (size_t)d * EE;
#pragma unroll
    for (int e = 0; e < EE; e++) p[e] += xv * g[e];
  }

  __shared__ float red[256][EE];
#pragma unroll
  for (int e = 0; e < EE; e++) red[threadIdx.x][e] = p[e];
  __syncthreads();

  if (threadIdx.x < EE) {
    int e = threadIdx.x;
    float s = 0.f;
    for (int i = 0; i < 256; i++) s += red[i][e];
    red[0][e] = s;
  }
  __syncthreads();

  if (threadIdx.x == 0) {
    float l[EE];
#pragma unroll
    for (int e = 0; e < EE; e++) l[e] = red[0][e];
    int e0 = 0;
#pragma unroll
    for (int e = 1; e < EE; e++) if (l[e] > l[e0]) e0 = e;   // first max wins (jax tie-break)
    int e1 = -1;
#pragma unroll
    for (int e = 0; e < EE; e++)
      if (e != e0 && (e1 < 0 || l[e] > l[e1])) e1 = e;
    float w1 = __expf(l[e1] - l[e0]);
    float inv = 1.f / (1.f + w1);
    pw[2 * t]     = inv;
    pw[2 * t + 1] = w1 * inv;
    int s0 = atomicAdd(&counts[e0], 1);
    lists[e0 * LISTCAP + s0] = 2 * t;
    int s1 = atomicAdd(&counts[e1], 1);
    lists[e1 * LISTCAP + s1] = 2 * t + 1;
  }
}

// ---------------------------------------------------------------- GEMM1: h = silu(x@wi0)*(x@wi1)*w
__global__ __launch_bounds__(256) void k_gemm1(
    const unsigned short* __restrict__ xbf,
    const float* __restrict__ wi0, const float* __restrict__ wi1,
    const float* __restrict__ pw, const int* __restrict__ counts,
    const int* __restrict__ lists, unsigned short* __restrict__ h) {
  const int e = blockIdx.z;
  const int cnt = counts[e];
  const int pBase = blockIdx.y * 128;
  if (pBase >= cnt) return;
  const int nBase = blockIdx.x * 128;

  __shared__ unsigned short Als[2][128][40];  // double-buffered A, async-filled
  __shared__ unsigned short B0s[128][40];     // [n][k]
  __shared__ unsigned short B1s[128][40];     // [n][k]
  __shared__ int   pArr[128];
  __shared__ float wArr[128];
  __shared__ int   tokArr[128];

  const int tid = threadIdx.x;
  if (tid < 128) {
    int idx = pBase + tid;
    int p = (idx < cnt) ? lists[e * LISTCAP + idx] : -1;
    pArr[tid]   = p;
    wArr[tid]   = (p >= 0) ? pw[p] : 0.f;
    tokArr[tid] = (p >= 0) ? (p >> 1) : 0;
  }
  __syncthreads();

  const int lane = tid & 31;
  const int wv = tid >> 5;      // 8 waves
  const int wr = wv & 3;        // *32 rows
  const int wc = wv >> 2;       // *64 cols
  const int lr = lane & 15;
  const int lg = lane >> 4;

  v8f acc0[2][4], acc1[2][4];
#pragma unroll
  for (int i = 0; i < 2; i++)
#pragma unroll
    for (int j = 0; j < 4; j++) { acc0[i][j] = {}; acc1[i][j] = {}; }

  // A async-copy mapping: 512 16B chunks; chunk c -> row c>>2, quarter c&3
  const int r0 = tid >> 2, q = tid & 3;
  const int r1 = 64 + r0;
  const unsigned short* abase0 = xbf + (size_t)tokArr[r0] * DD + q * 8;
  const unsigned short* abase1 = xbf + (size_t)tokArr[r1] * DD + q * 8;
  unsigned lA0[2], lA1[2];
#pragma unroll
  for (int b = 0; b < 2; b++) {
    lA0[b] = lds_off(&Als[b][r0][q * 8]);
    lA1[b] = lds_off(&Als[b][r1][q * 8]);
  }
  // prologue: stage k0=0 into buffer 0
  a_async16(lA0[0], abase0);
  a_async16(lA1[0], abase1);

  const size_t woff = (size_t)e * DD * MM;
  const int bn = tid & 127;
  const int bk = tid >> 7;      // 0..1

  for (int k0 = 0; k0 < DD; k0 += 32) {
    const int cur = (k0 >> 5) & 1, nxt = cur ^ 1;
    const bool more = (k0 + 32) < DD;

    const float* b0p = wi0 + woff + (size_t)(k0 + bk) * MM + (nBase + bn);
    const float* b1p = wi1 + woff + (size_t)(k0 + bk) * MM + (nBase + bn);
    float b0r[16], b1r[16];
#pragma unroll
    for (int kk = 0; kk < 16; kk++) {
      b0r[kk] = b0p[(size_t)kk * 2 * MM];
      b1r[kk] = b1p[(size_t)kk * 2 * MM];
    }
    __builtin_prefetch(b0p + (size_t)32 * MM, 0, 1);  // global_prefetch_b8
    __builtin_prefetch(b1p + (size_t)32 * MM, 0, 1);

    __syncthreads();   // all waves done reading previous-iteration LDS
    if (more) {        // async-fill the other A buffer; overlaps WMMA below
      a_async16(lA0[nxt], abase0 + k0 + 32);
      a_async16(lA1[nxt], abase1 + k0 + 32);
    }
#pragma unroll
    for (int kk = 0; kk < 16; kk++) {
      int k = bk + kk * 2;
      B0s[bn][k] = f2bf(b0r[kk]);
      B1s[bn][k] = f2bf(b1r[kk]);
    }
    // wait for current A buffer (older pair); async loads complete in order
    if (more) asm volatile("s_wait_asynccnt 0x2" ::: "memory");
    else      asm volatile("s_wait_asynccnt 0x0" ::: "memory");
    __syncthreads();

    Frag af[2];
#pragma unroll
    for (int i = 0; i < 2; i++) {
      int r = wr * 32 + i * 16 + lr;                 // ISA A layout: K = g*8.., 16+g*8..
      af[i].q[0] = *(const uint4*)&Als[cur][r][lg * 8];
      af[i].q[1] = *(const uint4*)&Als[cur][r][16 + lg * 8];
    }
    Frag b0f[4], b1f[4];
#pragma unroll
    for (int j = 0; j < 4; j++) {
      int n = wc * 64 + j * 16 + lr;                 // ISA B layout: K = g*8.., 16+g*8..
      b0f[j].q[0] = *(const uint4*)&B0s[n][lg * 8];
      b0f[j].q[1] = *(const uint4*)&B0s[n][16 + lg * 8];
      b1f[j].q[0] = *(const uint4*)&B1s[n][lg * 8];
      b1f[j].q[1] = *(const uint4*)&B1s[n][16 + lg * 8];
    }
#pragma unroll
    for (int i = 0; i < 2; i++)
#pragma unroll
      for (int j = 0; j < 4; j++) {
        acc0[i][j] = __builtin_amdgcn_wmma_f32_16x16x32_bf16(
            false, af[i].v, false, b0f[j].v, (short)0, acc0[i][j], false, false);
        acc1[i][j] = __builtin_amdgcn_wmma_f32_16x16x32_bf16(
            false, af[i].v, false, b1f[j].v, (short)0, acc1[i][j], false, false);
      }
  }

#pragma unroll
  for (int i = 0; i < 2; i++)
#pragma unroll
    for (int j = 0; j < 4; j++) {
      int nG = nBase + wc * 64 + j * 16 + lr;
#pragma unroll
      for (int jj = 0; jj < 8; jj++) {
        int m = wr * 32 + i * 16 + lg * 8 + jj;      // C/D layout: VGPR jj -> M = 8*g + jj
        int p = pArr[m];
        if (p >= 0) {
          float v0 = acc0[i][j][jj];
          float v1 = acc1[i][j][jj];
          float sig = __builtin_amdgcn_rcpf(1.f + __expf(-v0));  // v_rcp_f32, no div expansion
          float hv = v0 * sig * v1 * wArr[m];
          h[(size_t)p * MM + nG] = f2bf(hv);
        }
      }
    }
}

// ---------------------------------------------------------------- GEMM2: out_pairs = h @ wo[e]
__global__ __launch_bounds__(256) void k_gemm2(
    const unsigned short* __restrict__ h, const float* __restrict__ wo,
    const int* __restrict__ counts, const int* __restrict__ lists,
    float* __restrict__ outp) {
  const int e = blockIdx.z;
  const int cnt = counts[e];
  const int pBase = blockIdx.y * 128;
  if (pBase >= cnt) return;
  const int nBase = blockIdx.x * 128;

  __shared__ unsigned short Als[2][128][40];
  __shared__ unsigned short Bs[128][40];
  __shared__ int pArr[128];
  __shared__ int srcArr[128];

  const int tid = threadIdx.x;
  if (tid < 128) {
    int idx = pBase + tid;
    int p = (idx < cnt) ? lists[e * LISTCAP + idx] : -1;
    pArr[tid]   = p;
    srcArr[tid] = (p >= 0) ? p : 0;
  }
  __syncthreads();

  const int lane = tid & 31;
  const int wv = tid >> 5;
  const int wr = wv & 3;
  const int wc = wv >> 2;
  const int lr = lane & 15;
  const int lg = lane >> 4;

  v8f acc[2][4];
#pragma unroll
  for (int i = 0; i < 2; i++)
#pragma unroll
    for (int j = 0; j < 4; j++) acc[i][j] = {};

  const int r0 = tid >> 2, q = tid & 3;
  const int r1 = 64 + r0;
  const unsigned short* abase0 = h + (size_t)srcArr[r0] * MM + q * 8;
  const unsigned short* abase1 = h + (size_t)srcArr[r1] * MM + q * 8;
  unsigned lA0[2], lA1[2];
#pragma unroll
  for (int b = 0; b < 2; b++) {
    lA0[b] = lds_off(&Als[b][r0][q * 8]);
    lA1[b] = lds_off(&Als[b][r1][q * 8]);
  }
  a_async16(lA0[0], abase0);
  a_async16(lA1[0], abase1);

  const size_t woff = (size_t)e * MM * DD;
  const int bn = tid & 127;
  const int bk = tid >> 7;

  for (int k0 = 0; k0 < MM; k0 += 32) {
    const int cur = (k0 >> 5) & 1, nxt = cur ^ 1;
    const bool more = (k0 + 32) < MM;

    const float* bp = wo + woff + (size_t)(k0 + bk) * DD + (nBase + bn);
    float br[16];
#pragma unroll
    for (int kk = 0; kk < 16; kk++) br[kk] = bp[(size_t)kk * 2 * DD];
    __builtin_prefetch(bp + (size_t)32 * DD, 0, 1);

    __syncthreads();
    if (more) {
      a_async16(lA0[nxt], abase0 + k0 + 32);
      a_async16(lA1[nxt], abase1 + k0 + 32);
    }
#pragma unroll
    for (int kk = 0; kk < 16; kk++) Bs[bn][bk + kk * 2] = f2bf(br[kk]);
    if (more) asm volatile("s_wait_asynccnt 0x2" ::: "memory");
    else      asm volatile("s_wait_asynccnt 0x0" ::: "memory");
    __syncthreads();

    Frag af[2];
#pragma unroll
    for (int i = 0; i < 2; i++) {
      int r = wr * 32 + i * 16 + lr;
      af[i].q[0] = *(const uint4*)&Als[cur][r][lg * 8];
      af[i].q[1] = *(const uint4*)&Als[cur][r][16 + lg * 8];
    }
    Frag bf[4];
#pragma unroll
    for (int j = 0; j < 4; j++) {
      int n = wc * 64 + j * 16 + lr;
      bf[j].q[0] = *(const uint4*)&Bs[n][lg * 8];
      bf[j].q[1] = *(const uint4*)&Bs[n][16 + lg * 8];
    }
#pragma unroll
    for (int i = 0; i < 2; i++)
#pragma unroll
      for (int j = 0; j < 4; j++)
        acc[i][j] = __builtin_amdgcn_wmma_f32_16x16x32_bf16(
            false, af[i].v, false, bf[j].v, (short)0, acc[i][j], false, false);
  }

#pragma unroll
  for (int i = 0; i < 2; i++)
#pragma unroll
    for (int j = 0; j < 4; j++) {
      int nG = nBase + wc * 64 + j * 16 + lr;
#pragma unroll
      for (int jj = 0; jj < 8; jj++) {
        int m = wr * 32 + i * 16 + lg * 8 + jj;
        int p = pArr[m];
        if (p >= 0) outp[(size_t)p * DD + nG] = acc[i][j][jj];
      }
    }
}

// ---------------------------------------------------------------- combine the 2 expert outputs
__global__ void k_combine(const float* __restrict__ outp, float* __restrict__ out) {
  size_t i = (size_t)blockIdx.x * blockDim.x + threadIdx.x;
  size_t e4 = i * 4;
  size_t t = e4 >> 11;       // / DD
  size_t d = e4 & (DD - 1);
  const float4* a = (const float4*)(outp + (2 * t) * (size_t)DD + d);
  const float4* b = (const float4*)(outp + (2 * t + 1) * (size_t)DD + d);
  float4 av = *a, bv = *b;
  float4 r = make_float4(av.x + bv.x, av.y + bv.y, av.z + bv.z, av.w + bv.w);
  *(float4*)(out + e4) = r;
}

// ---------------------------------------------------------------- launch
extern "C" void kernel_launch(void* const* d_in, const int* in_sizes, int n_in,
                              void* d_out, int out_size, void* d_ws, size_t ws_size,
                              hipStream_t stream) {
  const float* x   = (const float*)d_in[0];
  const float* gk  = (const float*)d_in[1];
  const float* wi0 = (const float*)d_in[2];
  const float* wi1 = (const float*)d_in[3];
  const float* wo  = (const float*)d_in[4];
  float* out = (float*)d_out;

  char* ws = (char*)d_ws;
  // workspace layout (256B aligned)
  const size_t OFF_COUNTS = 0;                                   // 8 ints
  const size_t OFF_PW     = 4096;                                // 16384 f32
  const size_t OFF_LISTS  = OFF_PW + (size_t)NPAIR * 4;          // 8*16384 ints
  const size_t OFF_XBF    = OFF_LISTS + (size_t)EE * LISTCAP * 4;// 8192*2048 bf16
  const size_t OFF_H      = OFF_XBF + (size_t)NTOK * DD * 2;     // 16384*2048 bf16
  const size_t OFF_OP     = OFF_H + (size_t)NPAIR * MM * 2;      // 16384*2048 f32

  int*   counts = (int*)(ws + OFF_COUNTS);
  float* pw     = (float*)(ws + OFF_PW);
  int*   lists  = (int*)(ws + OFF_LISTS);
  unsigned short* xbf = (unsigned short*)(ws + OFF_XBF);
  unsigned short* hb  = (unsigned short*)(ws + OFF_H);
  float* outp   = (float*)(ws + OFF_OP);

  k_init<<<1, 32, 0, stream>>>(counts);
  k_gate<<<NTOK, 256, 0, stream>>>(x, gk, xbf, pw, counts, lists);
  k_gemm1<<<dim3(MM / 128, NPAIR / 128, EE), 256, 0, stream>>>(
      xbf, wi0, wi1, pw, counts, lists, hb);
  k_gemm2<<<dim3(DD / 128, NPAIR / 128, EE), 256, 0, stream>>>(
      hb, wo, counts, lists, outp);
  k_combine<<<(unsigned)(((size_t)NTOK * DD / 4) / 256), 256, 0, stream>>>(outp, out);
}
